// WaveRNN_18511309046344
// MI455X (gfx1250) — matrix operand order repaired and checked by reading, not compiled, not run on hardware
//
#include <hip/hip_runtime.h>
#include <hip/hip_bf16.h>
#include <math.h>

typedef __attribute__((ext_vector_type(16))) _Float16 v16h;
typedef __attribute__((ext_vector_type(8)))  _Float16 v8h;
typedef __attribute__((ext_vector_type(8)))  float    v8f;

__device__ __forceinline__ v16h cat16(v8h lo, v8h hi) {
    return __builtin_shufflevector(lo, hi, 0,1,2,3,4,5,6,7,8,9,10,11,12,13,14,15);
}

// Load one wave's A-fragment from a contiguous K run in LDS.
// Lane pattern (16-bit A, 16x32): lanes 0-15 hold K = {0..15},
// lanes 16-31 hold K = {16..31}. rowPtr points at K=0 of this lane's row.
__device__ __forceinline__ v16h ldsA(const _Float16* rowPtr, int half) {
    v8h lo = *(const v8h*)(rowPtr + half * 16);
    v8h hi = *(const v8h*)(rowPtr + half * 16 + 8);
    return cat16(lo, hi);
}

__device__ __forceinline__ v8f wmma_f16(v16h a, v16h b, v8f c) {
    return __builtin_amdgcn_wmma_f32_16x16x32_f16(false, a, false, b, (short)0, c, false, false);
}

__device__ __forceinline__ float sigmoidf_(float x) { return 1.f / (1.f + expf(-x)); }

// LDS layout (bytes), all 16B aligned:
//   sIn  : 8 x 2112 f16 = 33792   input, LDS[x] = in[x-16], zero-padded
//   sA1  : 8 x 2112 f16 = 33792   conv1 out, idx = pos*4 + ci + 32
//   sA2  : 8 x 1088 f16 = 17408   conv2 out, idx = pos*8 + ci + 24
//   sRnn : 16 x 32 f16  = 1024    GRU input rows (rows 8..15 zero)
//   sHid : 16 x 32 f16  = 1024    hidden rows    (rows 8..15 zero)
//   sGi  : 96 x 8 f32   = 3072
//   sGh  : 96 x 8 f32   = 3072
#define SMEM_BYTES 93184

__global__ void __launch_bounds__(256)
wavernn_fused(const float* __restrict__ past,
              const float* __restrict__ vel,
              const float* __restrict__ lpitch,
              const float* __restrict__ timef,
              const float* __restrict__ hidden,
              const float* __restrict__ w1, const float* __restrict__ b1,
              const float* __restrict__ w2, const float* __restrict__ b2,
              const float* __restrict__ w3, const float* __restrict__ b3,
              const float* __restrict__ w_ih, const float* __restrict__ w_hh,
              const float* __restrict__ b_ih, const float* __restrict__ b_hh,
              const float* __restrict__ w_proj, const float* __restrict__ b_proj,
              float* __restrict__ out, int B)
{
    extern __shared__ __align__(16) char smem[];
    _Float16* sIn  = (_Float16*)(smem);
    _Float16* sA1  = (_Float16*)(smem + 33792);
    _Float16* sA2  = (_Float16*)(smem + 67584);
    _Float16* sRnn = (_Float16*)(smem + 84992);
    _Float16* sHid = (_Float16*)(smem + 86016);
    float*    sGi  = (float*)(smem + 87040);
    float*    sGh  = (float*)(smem + 90112);

    const int tid  = threadIdx.x;
    const int wv   = tid >> 5;      // wave id 0..7 == local batch element
    const int lane = tid & 31;
    const int half = lane >> 4;
    const int l16  = lane & 15;
    const int gb   = blockIdx.x * 8 + wv;   // global batch element

    _Float16* in = sIn + wv * 2112;
    _Float16* a1 = sA1 + wv * 2112;
    _Float16* a2 = sA2 + wv * 1088;

    // ---------------- stage 0: zero LDS, stage input (f32 -> f16) ----------------
    for (int i = lane; i < 2112; i += 32) { in[i] = (_Float16)0.f; a1[i] = (_Float16)0.f; }
    for (int i = lane; i < 1088; i += 32) a2[i] = (_Float16)0.f;
    sRnn[wv * 32 + lane]       = (_Float16)0.f;
    sRnn[(wv + 8) * 32 + lane] = (_Float16)0.f;
    sHid[(wv + 8) * 32 + lane] = (_Float16)0.f;
    sHid[wv * 32 + lane]       = (_Float16)hidden[(long)gb * 32 + lane];

    const float4* p4 = (const float4*)(past + (long)gb * 2000);
    for (int i = lane; i < 500; i += 32) {
        float4 v = p4[i];
        int o = 16 + i * 4;
        in[o + 0] = (_Float16)v.x; in[o + 1] = (_Float16)v.y;
        in[o + 2] = (_Float16)v.z; in[o + 3] = (_Float16)v.w;
    }

    // ---------------- stage 1: conv1 (1->4ch, k=31, s=4) via WMMA ----------------
    // N column n = c*4 + s (4 out ch x 4 sub-positions), K j in [0,64):
    // output pos p = 4m + s reads in[16m - 16 + j], weight w1[c][j - 4s - 1].
    v16h b1f[2];
    {
        const int c = l16 >> 2, s = l16 & 3;
        #pragma unroll
        for (int q = 0; q < 2; ++q) {
            v16h f;
            #pragma unroll
            for (int i = 0; i < 16; ++i) {
                int j  = q * 32 + half * 16 + i;
                int kk = j - 4 * s - 1;
                f[i] = (_Float16)((kk >= 0 && kk < 31) ? w1[c * 31 + kk] : 0.f);
            }
            b1f[q] = f;
        }
    }
    {
        const float bias1 = b1[l16 >> 2];
        const int s = l16 & 3, c = l16 >> 2;
        #pragma unroll
        for (int t = 0; t < 8; ++t) {
            const _Float16* rp = in + 16 * (16 * t + l16);  // 16B aligned
            v8f acc = {};
            acc = wmma_f16(ldsA(rp +  0, half), b1f[0], acc);
            acc = wmma_f16(ldsA(rp + 32, half), b1f[1], acc);
            #pragma unroll
            for (int v = 0; v < 8; ++v) {
                int m = 16 * t + v + 8 * half;
                int pos = 4 * m + s;
                if (pos < 500)
                    a1[32 + pos * 4 + c] = (_Float16)fmaxf(acc[v] + bias1, 0.f);
            }
        }
    }

    // ---------------- stage 2: conv2 (4->8ch, k=15, s=4), two N-passes ----------------
    // pass P: n = c*2 + (s&1), s = 2P + (n&1); K j = rp*4 + ci in [0,128):
    // pos p = 4m + s reads element a1[(16m - 8 + rp)*4 + ci] (LDS idx 64m + j),
    // weight w2[c][ci][rp - 4s - 1].
    {
        const float bias2 = b2[l16 >> 1];
        #pragma unroll
        for (int P = 0; P < 2; ++P) {
            const int c = l16 >> 1, s = 2 * P + (l16 & 1);
            v16h b2f[4];
            #pragma unroll
            for (int q = 0; q < 4; ++q) {
                v16h f;
                #pragma unroll
                for (int i = 0; i < 16; ++i) {
                    int j   = q * 32 + half * 16 + i;
                    int rpj = j >> 2, ci = j & 3;
                    int kk  = rpj - 4 * s - 1;
                    f[i] = (_Float16)((kk >= 0 && kk < 15) ? w2[(c * 4 + ci) * 15 + kk] : 0.f);
                }
                b2f[q] = f;
            }
            #pragma unroll
            for (int t = 0; t < 2; ++t) {
                const _Float16* rp = a1 + 64 * (16 * t + l16);
                v8f acc = {};
                #pragma unroll
                for (int q = 0; q < 4; ++q)
                    acc = wmma_f16(ldsA(rp + 32 * q, half), b2f[q], acc);
                #pragma unroll
                for (int v = 0; v < 8; ++v) {
                    int m = 16 * t + v + 8 * half;
                    int pos = 4 * m + s;
                    if (pos < 125)
                        a2[24 + pos * 8 + c] = (_Float16)fmaxf(acc[v] + bias2, 0.f);
                }
            }
        }
    }

    // ---------------- stage 3: conv3 (8->16ch, k=7, s=4) + mean -> ctx ----------------
    // M = output position p directly (32 of them), N = 16 out channels,
    // K j = k*8 + ci in [0,64) (56 used): reads a2 element (4p-3+k)*8+ci (LDS idx 32p + j).
    {
        v16h b3f[2];
        #pragma unroll
        for (int q = 0; q < 2; ++q) {
            v16h f;
            #pragma unroll
            for (int i = 0; i < 16; ++i) {
                int j  = q * 32 + half * 16 + i;
                int kk = j >> 3, ci = j & 7;
                f[i] = (_Float16)((j < 56) ? w3[(l16 * 8 + ci) * 7 + kk] : 0.f);
            }
            b3f[q] = f;
        }
        const float bias3 = b3[l16];
        float csum = 0.f;
        #pragma unroll
        for (int t = 0; t < 2; ++t) {
            const _Float16* rp = a2 + 32 * (16 * t + l16);
            v8f acc = {};
            acc = wmma_f16(ldsA(rp +  0, half), b3f[0], acc);
            acc = wmma_f16(ldsA(rp + 32, half), b3f[1], acc);
            #pragma unroll
            for (int v = 0; v < 8; ++v) csum += fmaxf(acc[v] + bias3, 0.f);
        }
        csum += __shfl_xor(csum, 16, 32);               // combine both row-halves of column l16
        if (half == 0) sRnn[wv * 32 + l16] = (_Float16)(csum * (1.f / 32.f));
        if (lane < 3) {
            float f = (lane == 0) ? vel[gb] : (lane == 1) ? lpitch[gb] : timef[gb];
            sRnn[wv * 32 + 16 + lane] = (_Float16)f;
        }
    }
    __syncthreads();

    // ---------------- stage 4: GRU gates via WMMA (M=16 batch rows, N=96 gates) ----------------
    if (wv < 6) {
        const int g = 16 * wv + l16;                    // gate column 0..95
        v16h bih, bhh;
        #pragma unroll
        for (int i = 0; i < 16; ++i) {
            int k = half * 16 + i;
            bih[i] = (_Float16)((k < 19) ? w_ih[g * 19 + k] : 0.f);
            bhh[i] = (_Float16)w_hh[g * 32 + k];
        }
        v8f gi = {}, gh = {};
        gi = wmma_f16(ldsA(sRnn + 32 * l16, half), bih, gi);
        gh = wmma_f16(ldsA(sHid + 32 * l16, half), bhh, gh);
        const float bgi = b_ih[g], bgh = b_hh[g];
        #pragma unroll
        for (int v = 0; v < 8; ++v) {
            int row = v + 8 * half;                     // batch row
            if (row < 8) {
                sGi[g * 8 + row] = gi[v] + bgi;
                sGh[g * 8 + row] = gh[v] + bgh;
            }
        }
    }
    __syncthreads();

    // ---------------- stage 5: GRU nonlinearity + projection ----------------
    {
        const int b = wv;
        const int h = lane;                             // hidden dim 0..31
        float r = sigmoidf_(sGi[h * 8 + b]        + sGh[h * 8 + b]);
        float z = sigmoidf_(sGi[(32 + h) * 8 + b] + sGh[(32 + h) * 8 + b]);
        float n = tanhf(sGi[(64 + h) * 8 + b] + r * sGh[(64 + h) * 8 + b]);
        float hOld = hidden[(long)gb * 32 + h];
        float hNew = (1.f - z) * n + z * hOld;
        out[2 * (long)B + (long)gb * 32 + h] = hNew;

        float m0 = hNew * w_proj[h];
        float m1 = hNew * w_proj[32 + h];
        #pragma unroll
        for (int off = 16; off > 0; off >>= 1) {
            m0 += __shfl_xor(m0, off, 32);
            m1 += __shfl_xor(m1, off, 32);
        }
        if (lane == 0) {
            out[gb]     = m0 + b_proj[0];
            out[B + gb] = expf(m1 + b_proj[1]);
        }
    }
}

extern "C" void kernel_launch(void* const* d_in, const int* in_sizes, int n_in,
                              void* d_out, int out_size, void* d_ws, size_t ws_size,
                              hipStream_t stream) {
    const float* past   = (const float*)d_in[0];
    const float* vel    = (const float*)d_in[1];
    const float* lp     = (const float*)d_in[2];
    const float* tm     = (const float*)d_in[3];
    const float* hid    = (const float*)d_in[4];
    const float* w1     = (const float*)d_in[5];
    const float* b1     = (const float*)d_in[6];
    const float* w2     = (const float*)d_in[7];
    const float* b2     = (const float*)d_in[8];
    const float* w3     = (const float*)d_in[9];
    const float* b3     = (const float*)d_in[10];
    const float* w_ih   = (const float*)d_in[11];
    const float* w_hh   = (const float*)d_in[12];
    const float* b_ih   = (const float*)d_in[13];
    const float* b_hh   = (const float*)d_in[14];
    const float* w_proj = (const float*)d_in[15];
    const float* b_proj = (const float*)d_in[16];
    float* out = (float*)d_out;

    const int B = in_sizes[1];          // velocity is (B,)
    const int nBlocks = B / 8;          // 8 batch elements per workgroup

    static_assert(SMEM_BYTES <= 320 * 1024, "fits WGP LDS");
    hipFuncSetAttribute((const void*)wavernn_fused,
                        hipFuncAttributeMaxDynamicSharedMemorySize, SMEM_BYTES);

    wavernn_fused<<<nBlocks, 256, SMEM_BYTES, stream>>>(
        past, vel, lp, tm, hid, w1, b1, w2, b2, w3, b3,
        w_ih, w_hh, b_ih, b_hh, w_proj, b_proj, out, B);
}
